// SpikingBlock_10557029614258
// MI455X (gfx1250) — matrix operand order repaired and checked by reading, not compiled, not run on hardware
//
#include <hip/hip_runtime.h>

typedef __bf16 bf16_t;
typedef __attribute__((ext_vector_type(16))) __bf16 v16bf;
typedef __attribute__((ext_vector_type(8)))  __bf16 v8bf;
typedef __attribute__((ext_vector_type(4)))  __bf16 v4bf;
typedef __attribute__((ext_vector_type(8)))  float  v8f;

#define T_STEPS 64
#define BATCH   128
#define D_IN    2048
#define D_OUT   2048
#define M_DIM   (T_STEPS * BATCH)   /* 8192 */
#define K_DIM   D_IN                /* 2048 */
#define N_DIM   D_OUT               /* 2048 */
#define ALPHA   0.9f
#define THRESH  1.0f

#define BLK_M 128
#define BLK_N 128
#define BLK_K 32
#define LDA_P 40   /* padded pitch (halves) for A tile in LDS */
#define LDB_P 40   /* padded pitch (halves) for B^T tile in LDS */

// ---------------------------------------------------------------------------
// fp32 -> bf16 cast (vectorized: float4 in, 4x bf16 out)
// ---------------------------------------------------------------------------
__global__ void cast_f32_to_bf16_kernel(const float* __restrict__ src,
                                        bf16_t* __restrict__ dst, int n4) {
    int i = blockIdx.x * blockDim.x + threadIdx.x;
    if (i >= n4) return;
    float4 v = reinterpret_cast<const float4*>(src)[i];
    v4bf o;
    o[0] = (bf16_t)v.x; o[1] = (bf16_t)v.y;
    o[2] = (bf16_t)v.z; o[3] = (bf16_t)v.w;
    reinterpret_cast<v4bf*>(dst)[i] = o;
}

// ---------------------------------------------------------------------------
// X[M,N] = A[M,K](bf16) * W[K,N](bf16) + bias[N], fp32 accumulate via WMMA.
// 256 threads = 8 waves; block tile 128x128; wave tile 32x64 (2x4 WMMA tiles).
// ---------------------------------------------------------------------------
__global__ __launch_bounds__(256)
void gemm_bias_wmma_kernel(const bf16_t* __restrict__ A,
                           const bf16_t* __restrict__ W,
                           const float*  __restrict__ bias,
                           float* __restrict__ X) {
    __shared__ bf16_t Alds[BLK_M * LDA_P];          // [row][k], padded
    __shared__ bf16_t Blds[BLK_N * LDB_P];          // transposed: [n][k], padded

    const int tid  = threadIdx.x;
    const int lane = tid & 31;
    const int wave = tid >> 5;      // 0..7
    const int wr   = wave & 3;      // 4 row groups * 32 rows  = 128
    const int wc   = wave >> 2;     // 2 col groups * 64 cols  = 128

    const int rowBase = blockIdx.y * BLK_M;
    const int colBase = blockIdx.x * BLK_N;

    v8f acc[2][4];
#pragma unroll
    for (int mi = 0; mi < 2; ++mi)
#pragma unroll
        for (int ni = 0; ni < 4; ++ni)
#pragma unroll
            for (int v = 0; v < 8; ++v) acc[mi][ni][v] = 0.0f;

    // Global->LDS staging indices
    const int lr  = tid >> 1;             // A: row within tile (0..127)
    const int lk  = (tid & 1) * 16;       // A: k offset (0 or 16)
    const int bk  = tid >> 4;             // B: k row (0..15) + pass*16
    const int bn0 = (tid & 15) * 8;       // B: n offset (0..120 step 8)

    // Fragment indices (ISA 7.12.2 16-bit layouts, wave32)
    const int kb   = lane >> 4;           // lane group 0/1
    const int lr16 = lane & 15;

    for (int kt = 0; kt < K_DIM; kt += BLK_K) {
        // ---- stage A tile 128x32 (coalesced 16B loads) ----
        const v8bf* asrc = reinterpret_cast<const v8bf*>(
            &A[(size_t)(rowBase + lr) * K_DIM + kt + lk]);
        v8bf a0 = asrc[0];
        v8bf a1 = asrc[1];
        *reinterpret_cast<v8bf*>(&Alds[lr * LDA_P + lk])     = a0;
        *reinterpret_cast<v8bf*>(&Alds[lr * LDA_P + lk + 8]) = a1;

        // ---- stage B tile 32x128, transposed into LDS as [n][k] ----
#pragma unroll
        for (int p = 0; p < 2; ++p) {
            const int k = bk + p * 16;
            v8bf wv = *reinterpret_cast<const v8bf*>(
                &W[(size_t)(kt + k) * N_DIM + colBase + bn0]);
#pragma unroll
            for (int i = 0; i < 8; ++i)
                Blds[(bn0 + i) * LDB_P + k] = wv[i];
        }
        __syncthreads();

        // ---- build A fragments: lane<16 -> K {kb*8..+8, 16+kb*8..+8} ----
        v16bf afrag[2];
#pragma unroll
        for (int mi = 0; mi < 2; ++mi) {
            const bf16_t* ap = &Alds[(wr * 32 + mi * 16 + lr16) * LDA_P];
            v8bf lo = *reinterpret_cast<const v8bf*>(ap + kb * 8);
            v8bf hi = *reinterpret_cast<const v8bf*>(ap + 16 + kb * 8);
            afrag[mi] = __builtin_shufflevector(lo, hi,
                0,1,2,3,4,5,6,7,8,9,10,11,12,13,14,15);
        }
        // ---- build B fragments: col = lane%16, K {kb*8..+8, 16+kb*8..+8} ----
        v16bf bfrag[4];
#pragma unroll
        for (int ni = 0; ni < 4; ++ni) {
            const bf16_t* bp = &Blds[(wc * 64 + ni * 16 + lr16) * LDB_P + kb * 8];
            v8bf lo = *reinterpret_cast<const v8bf*>(bp);
            v8bf hi = *reinterpret_cast<const v8bf*>(bp + 16);
            bfrag[ni] = __builtin_shufflevector(lo, hi,
                0,1,2,3,4,5,6,7,8,9,10,11,12,13,14,15);
        }

        // ---- 8x v_wmma_f32_16x16x32_bf16 per wave per K-step ----
#pragma unroll
        for (int mi = 0; mi < 2; ++mi)
#pragma unroll
            for (int ni = 0; ni < 4; ++ni)
                acc[mi][ni] = __builtin_amdgcn_wmma_f32_16x16x32_bf16(
                    /*neg_a=*/false, afrag[mi],
                    /*neg_b=*/false, bfrag[ni],
                    /*c_mod=*/(short)0, acc[mi][ni],
                    /*reuse_a=*/false, /*reuse_b=*/false);
        __syncthreads();
    }

    // ---- epilogue: D layout VGPR v -> M = v + 8*(lane/16), N = lane%16 ----
#pragma unroll
    for (int mi = 0; mi < 2; ++mi) {
#pragma unroll
        for (int ni = 0; ni < 4; ++ni) {
            const int col  = colBase + wc * 64 + ni * 16 + lr16;
            const float bb = bias[col];
            const int row0 = rowBase + wr * 32 + mi * 16 + kb * 8;
#pragma unroll
            for (int v = 0; v < 8; ++v)
                X[(size_t)(row0 + v) * N_DIM + col] = acc[mi][ni][v] + bb;
        }
    }
}

// ---------------------------------------------------------------------------
// LIF scan: one thread per (b, n). Reads X (pre-activations) in-place from the
// s_seq region and overwrites with spikes; writes u_final to out[0 .. B*D).
// ---------------------------------------------------------------------------
__global__ __launch_bounds__(256)
void lif_scan_kernel(const float* __restrict__ u0,
                     float* __restrict__ Xspk,   // [T, B*D] in-place X -> spikes
                     float* __restrict__ ufin) { // [B*D]
    const int idx = blockIdx.x * blockDim.x + threadIdx.x;
    const int BD  = BATCH * D_OUT;
    float u = u0[idx];
#pragma unroll 4
    for (int t = 0; t < T_STEPS; ++t) {
        const float x = Xspk[(size_t)t * BD + idx];
        u = ALPHA * u + x;
        const float s = (u >= THRESH) ? 1.0f : 0.0f;
        Xspk[(size_t)t * BD + idx] = s;
        u -= s * THRESH;
    }
    ufin[idx] = u;
}

// ---------------------------------------------------------------------------
extern "C" void kernel_launch(void* const* d_in, const int* in_sizes, int n_in,
                              void* d_out, int out_size, void* d_ws, size_t ws_size,
                              hipStream_t stream) {
    (void)in_sizes; (void)n_in; (void)out_size; (void)ws_size;

    const float* inputs = (const float*)d_in[0];  // [T, B, D_in]
    const float* u0     = (const float*)d_in[1];  // [B, D_out]
    const float* W      = (const float*)d_in[2];  // [D_in, D_out]
    const float* bias   = (const float*)d_in[3];  // [D_out]
    float* out = (float*)d_out;                   // u_final | s_seq

    // Workspace: bf16 casts of A and W
    bf16_t* Abf = (bf16_t*)d_ws;
    bf16_t* Wbf = (bf16_t*)((char*)d_ws + (size_t)M_DIM * K_DIM * sizeof(bf16_t));

    const int nA4 = (M_DIM * K_DIM) / 4;  // 4,194,304
    const int nW4 = (K_DIM * N_DIM) / 4;  // 1,048,576
    cast_f32_to_bf16_kernel<<<(nA4 + 255) / 256, 256, 0, stream>>>(inputs, Abf, nA4);
    cast_f32_to_bf16_kernel<<<(nW4 + 255) / 256, 256, 0, stream>>>(W, Wbf, nW4);

    // GEMM writes pre-activations X directly into the s_seq region of d_out.
    float* X = out + (size_t)BATCH * D_OUT;
    dim3 grid(N_DIM / BLK_N, M_DIM / BLK_M);   // (16, 64)
    gemm_bias_wmma_kernel<<<grid, 256, 0, stream>>>(Abf, Wbf, bias, X);

    // Sequential LIF dynamics; overwrites X with spikes, emits u_final.
    lif_scan_kernel<<<(BATCH * D_OUT) / 256, 256, 0, stream>>>(u0, X, out);
}
